// NetWith4BitLinear_16449724744367
// MI455X (gfx1250) — matrix operand order repaired and checked by reading, not compiled, not run on hardware
//
#include <hip/hip_runtime.h>
#include <hip/hip_bf16.h>

typedef __bf16 bf16;
typedef __attribute__((ext_vector_type(16))) __bf16 v16bf;
typedef __attribute__((ext_vector_type(8)))  __bf16 v8bf;
typedef __attribute__((ext_vector_type(4)))  __bf16 v4bf;
typedef __attribute__((ext_vector_type(8)))  float  v8f;
typedef __attribute__((ext_vector_type(4)))  float  v4f;
typedef __attribute__((ext_vector_type(4)))  int    v4i;

#define BM 256
#define BN 128
#define BK 64
#define NT 512    // threads per block (16 wave32s, 8x2 wave grid of 32x64 tiles)
#define LDSP 72   // padded LDS row stride (elements): 144B breaks ds_load_b128 conflicts

// bitsandbytes FP4 codebook (sign bit = code & 8)
__device__ __constant__ float FP4_TBL[16] = {
     0.0f,  0.0052083333f,  0.6666667f,  1.0f,  0.3333333f,  0.5f,  0.1666667f,  0.25f,
    -0.0f, -0.0052083333f, -0.6666667f, -1.0f, -0.3333333f, -0.5f, -0.1666667f, -0.25f
};

// f32 -> bf16 bulk conversion (layer-1 activations), 8 elems/thread
__global__ __launch_bounds__(256)
void cvt_f32_bf16(const float* __restrict__ in, bf16* __restrict__ out)
{
    size_t i = ((size_t)blockIdx.x * 256 + threadIdx.x) * 8;
    v4f a = *(const v4f*)(in + i);
    v4f b = *(const v4f*)(in + i + 4);
    v8bf o;
    o[0] = (bf16)a.x; o[1] = (bf16)a.y; o[2] = (bf16)a.z; o[3] = (bf16)a.w;
    o[4] = (bf16)b.x; o[5] = (bf16)b.y; o[6] = (bf16)b.z; o[7] = (bf16)b.w;
    *(v8bf*)(out + i) = o;
}

// out[M,N] = A[M,K] (bf16 compute) @ dequant(codes[N,K])^T + bias[N], optional ReLU.
// Software-pipelined: global loads for stage s+1 issue before the barrier,
// WMMA for stage s runs from LDS buffer s&1, dequant+store fills buffer (s+1)&1.
template <typename AT, bool RELU>
__global__ __launch_bounds__(NT)
void gemm_fp4(const AT* __restrict__ A, const int* __restrict__ codes,
              const float* __restrict__ absmax, const float* __restrict__ bias,
              bf16* __restrict__ out, int M, int N, int K)
{
    __shared__ __align__(16) bf16 As[2][BM * LDSP];
    __shared__ __align__(16) bf16 Ws[2][BN * LDSP];
    __shared__ float tbl[16];

    const int tid   = threadIdx.x;
    const int lane  = tid & 31;
    const int wave  = tid >> 5;     // 0..15
    const int wx    = wave & 1;     // 0..1 : 64-col slab
    const int wy    = wave >> 1;    // 0..7 : 32-row slab
    const int lrow  = lane & 15;
    const int lhalf = lane >> 4;

    const int m0 = blockIdx.y * BM;
    const int n0 = blockIdx.x * BN;
    const int kblocks = K >> 6;     // absmax blocks per weight row (BLOCK=64)

    if (tid < 16) tbl[tid] = FP4_TBL[tid];

    // ---- staging registers (live across the compute phase) ----
    v4f   af[8];   // f32 activation path: BM*BK/4/NT = 8 chunks
    v8bf  ab[4];   // bf16 activation path: BM*BK/8/NT = 4 chunks
    v4i   qv[4];   // weight codes:        BN*BK/4/NT = 4 chunks
    float sv[4];   // absmax scales

    auto gload = [&](int k0) {
        if constexpr (sizeof(AT) == 4) {
#pragma unroll
            for (int i = 0; i < 8; ++i) {
                int c = tid + NT * i;
                int r = c >> 4, col = (c & 15) * 4;
                af[i] = *(const v4f*)(A + (size_t)(m0 + r) * K + k0 + col);
            }
        } else {
#pragma unroll
            for (int i = 0; i < 4; ++i) {
                int c = tid + NT * i;
                int r = c >> 3, col = (c & 7) * 8;
                ab[i] = *(const v8bf*)(A + (size_t)(m0 + r) * K + k0 + col);
            }
        }
#pragma unroll
        for (int i = 0; i < 4; ++i) {
            int c = tid + NT * i;
            int r = c >> 4, col = (c & 15) * 4;
            qv[i] = *(const v4i*)(codes + (size_t)(n0 + r) * K + k0 + col);
            sv[i] = absmax[(size_t)(n0 + r) * kblocks + (k0 >> 6)];
        }
    };

    auto lstore = [&](int buf) {
        bf16* Asb = As[buf];
        bf16* Wsb = Ws[buf];
        if constexpr (sizeof(AT) == 4) {
#pragma unroll
            for (int i = 0; i < 8; ++i) {
                int c = tid + NT * i;
                int r = c >> 4, col = (c & 15) * 4;
                v4bf b;
                b.x = (bf16)af[i].x; b.y = (bf16)af[i].y;
                b.z = (bf16)af[i].z; b.w = (bf16)af[i].w;
                *(v4bf*)(Asb + r * LDSP + col) = b;
            }
        } else {
#pragma unroll
            for (int i = 0; i < 4; ++i) {
                int c = tid + NT * i;
                int r = c >> 3, col = (c & 7) * 8;
                *(v8bf*)(Asb + r * LDSP + col) = ab[i];
            }
        }
#pragma unroll
        for (int i = 0; i < 4; ++i) {
            int c = tid + NT * i;
            int r = c >> 4, col = (c & 15) * 4;
            float s = sv[i];
            v4bf b;
            b.x = (bf16)(tbl[qv[i].x & 15] * s);
            b.y = (bf16)(tbl[qv[i].y & 15] * s);
            b.z = (bf16)(tbl[qv[i].z & 15] * s);
            b.w = (bf16)(tbl[qv[i].w & 15] * s);
            *(v4bf*)(Wsb + r * LDSP + col) = b;
        }
    };

    v8f acc[2][4];
#pragma unroll
    for (int mi = 0; mi < 2; ++mi)
#pragma unroll
        for (int ni = 0; ni < 4; ++ni)
#pragma unroll
            for (int j = 0; j < 8; ++j) acc[mi][ni][j] = 0.0f;

    const int S = K / BK;

    // prologue: stage 0 into buffer 0
    gload(0);
    __syncthreads();          // tbl visible to all waves
    lstore(0);

    for (int s = 0; s < S; ++s) {
        if (s + 1 < S) gload((s + 1) * BK);   // long-latency loads, hidden by WMMA below

        __syncthreads();      // buf[s&1] published; all waves done reading buf[(s+1)&1]

        const bf16* Asb = As[s & 1];
        const bf16* Wsb = Ws[s & 1];
#pragma unroll
        for (int kk = 0; kk < BK; kk += 32) {
            v16bf a[2], b[4];
#pragma unroll
            for (int mi = 0; mi < 2; ++mi) {
                // A 16x32 bf16 frag: lanes0-15 K[0..7],K[16..23]; lanes16-31 K[8..15],K[24..31]
                const bf16* p = Asb + (wy * 32 + mi * 16 + lrow) * LDSP + kk + 8 * lhalf;
                union { v16bf v; v8bf h[2]; } u;
                u.h[0] = *(const v8bf*)p;
                u.h[1] = *(const v8bf*)(p + 16);
                a[mi] = u.v;
            }
#pragma unroll
            for (int ni = 0; ni < 4; ++ni) {
                // B 32x16 frag: lane = column n, lanes0-15 K[0..7],K[16..23]; lanes16-31 K[8..15],K[24..31]
                const bf16* p = Wsb + (wx * 64 + ni * 16 + lrow) * LDSP + kk + 8 * lhalf;
                union { v16bf v; v8bf h[2]; } u;
                u.h[0] = *(const v8bf*)p;
                u.h[1] = *(const v8bf*)(p + 16);
                b[ni] = u.v;
            }
#pragma unroll
            for (int mi = 0; mi < 2; ++mi)
#pragma unroll
                for (int ni = 0; ni < 4; ++ni)
                    acc[mi][ni] = __builtin_amdgcn_wmma_f32_16x16x32_bf16(
                        false, a[mi], false, b[ni], (short)0, acc[mi][ni], false, false);
        }

        if (s + 1 < S) lstore((s + 1) & 1);   // overlaps other waves' WMMAs
    }

    // ---- epilogue: bias (+ReLU) and bf16 store ----
    // C/D layout: VGPR r -> row 8*lhalf + r, col = lane&15 within each 16x16 tile
#pragma unroll
    for (int ni = 0; ni < 4; ++ni) {
        int n = n0 + wx * 64 + ni * 16 + lrow;
        float bv = bias[n];
#pragma unroll
        for (int mi = 0; mi < 2; ++mi) {
#pragma unroll
            for (int r = 0; r < 8; ++r) {
                int m = m0 + wy * 32 + mi * 16 + 8 * lhalf + r;
                float v = acc[mi][ni][r] + bv;
                if (RELU) v = v > 0.0f ? v : 0.0f;
                out[(size_t)m * N + n] = (bf16)v;
            }
        }
    }
}

extern "C" void kernel_launch(void* const* d_in, const int* in_sizes, int n_in,
                              void* d_out, int out_size, void* d_ws, size_t ws_size,
                              hipStream_t stream)
{
    const int B = 16384, I = 2048, H = 4096, O = 2048;

    const float* x   = (const float*)d_in[0];
    const int*   c1  = (const int*)  d_in[1];
    const float* am1 = (const float*)d_in[2];
    const float* b1  = (const float*)d_in[3];
    const int*   c2  = (const int*)  d_in[4];
    const float* am2 = (const float*)d_in[5];
    const float* b2  = (const float*)d_in[6];
    const int*   c3  = (const int*)  d_in[7];
    const float* am3 = (const float*)d_in[8];
    const float* b3  = (const float*)d_in[9];

    bf16* h1 = (bf16*)d_ws;                      // [B,H] bf16 = 128 MiB
    bf16* h2 = h1 + (size_t)B * H;               // [B,H] bf16 = 128 MiB
    bf16* xb = h2 + (size_t)B * H;               // [B,I] bf16 =  64 MiB (optional)
    bf16* y  = (bf16*)d_out;                     // [B,O] bf16

    size_t need = (2 * (size_t)B * H + (size_t)B * I) * sizeof(bf16);
    bool pre_cvt = ws_size >= need;

    dim3 blk(NT);
    if (pre_cvt) {
        // convert x once; all three GEMMs use the lighter bf16 activation path
        cvt_f32_bf16<<<dim3(((size_t)B * I) / (256 * 8)), dim3(256), 0, stream>>>(x, xb);
        gemm_fp4<bf16, true ><<<dim3(H / BN, B / BM), blk, 0, stream>>>(xb, c1, am1, b1, h1, B, H, I);
    } else {
        gemm_fp4<float, true ><<<dim3(H / BN, B / BM), blk, 0, stream>>>(x, c1, am1, b1, h1, B, H, I);
    }
    gemm_fp4<bf16, true ><<<dim3(H / BN, B / BM), blk, 0, stream>>>(h1, c2, am2, b2, h2, B, H, H);
    gemm_fp4<bf16, false><<<dim3(O / BN, B / BM), blk, 0, stream>>>(h2, c3, am3, b3, y,  B, O, H);
}
